// Block_41609643164089
// MI455X (gfx1250) — compile-verified, matrix operand order checked by reading, not run on hardware
//
#include <hip/hip_runtime.h>
#include <hip/hip_bf16.h>
#include <hip/hip_fp16.h>

// ---------------------------------------------------------------------------
// Mamba-style block for MI455X (gfx1250, wave32, WMMA + Tensor Data Mover).
//   GEMMs: f16 inputs, f32 accumulation via v_wmma_f32_16x16x32_f16.
//          A/B tiles staged into LDS by TDM (tensor_load_to_lds), double
//          buffered, consumed as ds_load_b128 fragments.
//   SSM scan: 3-pass chunked parallel scan, state in VGPRs, B/C staged in LDS.
// ---------------------------------------------------------------------------

typedef __attribute__((ext_vector_type(16))) _Float16     v16h;
typedef __attribute__((ext_vector_type(8)))  float        v8f;
typedef __attribute__((ext_vector_type(4)))  unsigned int u32x4;
typedef __attribute__((ext_vector_type(8)))  int          i32x8;
typedef __attribute__((ext_vector_type(4)))  int          i32x4;

#define B_  4
#define L_  1024
#define D_  1024
#define N_  16
#define R_  64
#define E_  96                // R + 2N
#define F_  4096              // 4*D
#define M_  (B_ * L_)         // 4096 token rows
#define NCHUNK 8
#define LCHUNK (L_ / NCHUNK)  // 128

#if defined(__has_builtin)
#if __has_builtin(__builtin_amdgcn_tensor_load_to_lds) && \
    __has_builtin(__builtin_amdgcn_s_wait_tensorcnt)
#define USE_TDM 1
#endif
#endif
#ifndef USE_TDM
#define USE_TDM 0
#endif

// ------------------------------- helpers -----------------------------------

__device__ __forceinline__ float softplusf(float x) {
    return x > 20.f ? x : log1pf(__expf(x));
}

__device__ __forceinline__ float geluf(float x) {
    float x3 = x * x * x;
    return 0.5f * x * (1.f + tanhf(0.7978845608028654f * (x + 0.044715f * x3)));
}

// 16x32 f16 WMMA fragment for one lane from a K-contiguous row-major matrix.
// ISA layout (16-bit A-matrix 16x32): lane&15 selects the row, lane>>4 selects
// K-halves {0..7,16..23} vs {8..15,24..31}  ->  two b128 loads.
__device__ __forceinline__ v16h load_frag(const _Float16* __restrict__ base,
                                          int ld, int row, int k0, int lane) {
    const _Float16* p = base + (size_t)row * ld + k0 + ((lane >> 4) & 1) * 8;
    union { v16h v; uint4 q[2]; } f;
    f.q[0] = *reinterpret_cast<const uint4*>(p);
    f.q[1] = *reinterpret_cast<const uint4*>(p + 16);
    return f.v;
}

// Same fragment pattern, but from a [rows][32] f16 tile in LDS -> ds_load_b128.
__device__ __forceinline__ v16h lds_frag(const _Float16* tile, int row, int lane) {
    const _Float16* p = tile + row * 32 + ((lane >> 4) & 1) * 8;
    union { v16h v; uint4 q[2]; } f;
    f.q[0] = *reinterpret_cast<const uint4*>(p);
    f.q[1] = *reinterpret_cast<const uint4*>(p + 16);
    return f.v;
}

#if USE_TDM
// Issue one TDM 2D tile load: 128 rows x 32 f16 cols from a row-major matrix
// (ld_elems f16 per row) into LDS.  Rows beyond rows_remaining read as zero
// (TDM OOB policy).  D# packing per CDNA5 ISA ch.8.  This toolchain declares
// the 6-operand builtin: (g0 u32x4, g1 i32x8, g2 i32x4, g3 i32x4, i32x8, cpol).
__device__ __forceinline__ void tdm_load_tile_f16(_Float16* lds_dst,
                                                  const _Float16* gsrc,
                                                  int ld_elems,
                                                  int rows_remaining) {
    uint64_t ga     = (uint64_t)(uintptr_t)gsrc;
    uint32_t ldsoff = (uint32_t)(uintptr_t)(void*)lds_dst;  // low 32b = LDS addr

    u32x4 g0;
    g0.x = 1u;                                   // count=1 (valid, user mode)
    g0.y = ldsoff;                               // lds_addr
    g0.z = (uint32_t)ga;                         // global_addr[31:0]
    g0.w = ((uint32_t)(ga >> 32) & 0x01FFFFFFu)  // global_addr[56:32]
         | (2u << 30);                           // type = 2 ("image")

    uint64_t q0 = 0x10000ull;                                        // data_size=1 (2B)
    q0 |= ((uint64_t)((uint32_t)ld_elems & 0xFFFFu)) << 48;          // tensor_dim0[15:0]
    uint64_t q1 = (uint64_t)(((uint32_t)ld_elems >> 16) & 0xFFFFu);  // tensor_dim0[31:16]
    q1 |= ((uint64_t)(uint32_t)rows_remaining) << 16;                // tensor_dim1
    q1 |= 32ull << 48;                                               // tile_dim0 = 32
    uint64_t q2 = 128ull;                                            // tile_dim1 = 128
    q2 |= ((uint64_t)(uint32_t)ld_elems) << 32;                      // dim0_stride[31:0]
    uint64_t q3 = 0;                                                 // stride msbs / dim1_stride

    union { uint64_t q[4]; i32x8 v; } g1u;
    g1u.q[0] = q0; g1u.q[1] = q1; g1u.q[2] = q2; g1u.q[3] = q3;
    i32x4 z4 = {};
    i32x8 z8 = {};
    __builtin_amdgcn_tensor_load_to_lds(g0, g1u.v, z4, z4, z8, 0);
}
#endif

// ------------------------------- kernels -----------------------------------

__global__ __launch_bounds__(256) void cvt_f16(const float* __restrict__ s,
                                               _Float16* __restrict__ d, int n) {
    int i = blockIdx.x * 256 + threadIdx.x;
    if (i < n) d[i] = (_Float16)s[i];
}

// MODE 0: ln1 + softplus -> f32 + f16 copies.  MODE 1: ln2 -> f16 only.
template <int MODE>
__global__ __launch_bounds__(256) void ln_kernel(const float* __restrict__ x,
                                                 const float* __restrict__ w,
                                                 float* __restrict__ of32,
                                                 _Float16* __restrict__ of16) {
    __shared__ float s1[256], s2[256];
    const int row = blockIdx.x;
    const int tid = threadIdx.x;
    const float* xr = x + (size_t)row * D_;
    float v[4], a = 0.f, b = 0.f;
#pragma unroll
    for (int j = 0; j < 4; j++) {
        v[j] = xr[tid + j * 256];
        a += v[j];
        b += v[j] * v[j];
    }
    s1[tid] = a; s2[tid] = b;
    __syncthreads();
    for (int off = 128; off > 0; off >>= 1) {
        if (tid < off) { s1[tid] += s1[tid + off]; s2[tid] += s2[tid + off]; }
        __syncthreads();
    }
    const float mu  = s1[0] * (1.f / D_);
    const float var = s2[0] * (1.f / D_) - mu * mu;
    const float rs  = rsqrtf(var + 1e-5f);
#pragma unroll
    for (int j = 0; j < 4; j++) {
        int c = tid + j * 256;
        float y = (v[j] - mu) * rs * w[c];
        if (MODE == 0) {
            y = softplusf(y);
            of32[(size_t)row * D_ + c] = y;
            of16[(size_t)row * D_ + c] = (_Float16)y;
        } else {
            of16[(size_t)row * D_ + c] = (_Float16)y;
        }
    }
}

// Generic WMMA GEMM: out[m,n] = sum_k A[m,k] * W[n,k]  (A:(M,lda), W:(N,ldb)).
// Block = 8 waves; wave tile = 32(M) x 64(N) = 8 accumulators; block tile 128x128.
// TDM path: wave0 DMAs the 128x32 A and B tiles into double-buffered LDS while
// all waves compute on the previous pair; ds_load_b128 fragment reads.
// EPI: 0 = dual f32+f16 store (dbc), 1 = +bias & softplus (delta),
//      2 = gelu -> f16 (fc),         3 = +resid -> f32 (proj/out).
template <int EPI>
__global__ __launch_bounds__(256) void gemm_wmma(
    const _Float16* __restrict__ A, int lda,
    const _Float16* __restrict__ Bw, int ldb,
    int M, int N, int K,
    const float* __restrict__ bias, const float* __restrict__ resid,
    float* __restrict__ outF, _Float16* __restrict__ outH, int ldo) {
    const int lane  = threadIdx.x & 31;
    const int wave  = threadIdx.x >> 5;
    const int waveM = wave >> 1;  // 0..3
    const int waveN = wave & 1;   // 0..1
    const int tileM = blockIdx.y * 128 + waveM * 32;
    const int tileN = blockIdx.x * 128 + waveN * 64;
    const int col16 = lane & 15;
    const int hi8   = (lane >> 4) * 8;

    v8f acc[2][4] = {};

#if USE_TDM
    __shared__ __align__(16) _Float16 at[2][128 * 32];
    __shared__ __align__(16) _Float16 bt[2][128 * 32];
    const int rowbase = blockIdx.y * 128;
    const int colbase = blockIdx.x * 128;
    if (threadIdx.x < 32) {   // wave 0 drives the Tensor Data Mover
        tdm_load_tile_f16(at[0], A  + (size_t)rowbase * lda, lda, M - rowbase);
        tdm_load_tile_f16(bt[0], Bw + (size_t)colbase * ldb, ldb, N - colbase);
    }
    int cur = 0;
    for (int k0 = 0; k0 < K; k0 += 32) {
        if (threadIdx.x < 32) __builtin_amdgcn_s_wait_tensorcnt(0);
        __syncthreads();      // tile(k0) visible to all waves; prev compute done
        if (threadIdx.x < 32 && k0 + 32 < K) {   // prefetch tile(k0+32) via TDM
            tdm_load_tile_f16(at[cur ^ 1], A  + (size_t)rowbase * lda + k0 + 32,
                              lda, M - rowbase);
            tdm_load_tile_f16(bt[cur ^ 1], Bw + (size_t)colbase * ldb + k0 + 32,
                              ldb, N - colbase);
        }
        v16h a0 = lds_frag(at[cur], waveM * 32 + col16, lane);
        v16h a1 = lds_frag(at[cur], waveM * 32 + 16 + col16, lane);
#pragma unroll
        for (int nt = 0; nt < 4; nt++) {
            v16h bfr = lds_frag(bt[cur], waveN * 64 + nt * 16 + col16, lane);
            acc[0][nt] = __builtin_amdgcn_wmma_f32_16x16x32_f16(
                false, a0, false, bfr, (short)0, acc[0][nt], false, false);
            acc[1][nt] = __builtin_amdgcn_wmma_f32_16x16x32_f16(
                false, a1, false, bfr, (short)0, acc[1][nt], false, false);
        }
        cur ^= 1;
    }
#else
    int bro[4];
#pragma unroll
    for (int nt = 0; nt < 4; nt++) {        // clamp OOB weight rows (dbc N=96)
        int c = tileN + nt * 16 + col16;
        bro[nt] = c < N ? c : N - 1;
    }
    const int aro0 = tileM + col16;
    const int aro1 = tileM + 16 + col16;
    for (int k0 = 0; k0 < K; k0 += 32) {
        if (k0 + 32 < K) {                  // gfx1250 global_prefetch_b8
            __builtin_prefetch(A  + (size_t)aro0   * lda + k0 + 32, 0, 1);
            __builtin_prefetch(Bw + (size_t)bro[0] * ldb + k0 + 32, 0, 1);
        }
        v16h a0 = load_frag(A, lda, aro0, k0, lane);
        v16h a1 = load_frag(A, lda, aro1, k0, lane);
#pragma unroll
        for (int nt = 0; nt < 4; nt++) {
            v16h bfr = load_frag(Bw, ldb, bro[nt], k0, lane);
            acc[0][nt] = __builtin_amdgcn_wmma_f32_16x16x32_f16(
                false, a0, false, bfr, (short)0, acc[0][nt], false, false);
            acc[1][nt] = __builtin_amdgcn_wmma_f32_16x16x32_f16(
                false, a1, false, bfr, (short)0, acc[1][nt], false, false);
        }
    }
#endif

#pragma unroll
    for (int mt = 0; mt < 2; mt++) {
#pragma unroll
        for (int nt = 0; nt < 4; nt++) {
            int col = tileN + nt * 16 + col16;
            if (col >= N) continue;
#pragma unroll
            for (int r = 0; r < 8; r++) {   // C/D layout: VGPR r -> M = r + hi8
                int row  = tileM + mt * 16 + hi8 + r;
                float v  = acc[mt][nt][r];
                size_t o = (size_t)row * ldo + col;
                if (EPI == 0)      { outF[o] = v; outH[o] = (_Float16)v; }
                else if (EPI == 1) { outF[o] = softplusf(v + bias[col]); }
                else if (EPI == 2) { outH[o] = (_Float16)geluf(v); }
                else               { outF[o] = resid[o] + v; }
            }
        }
    }
}

// Chunked SSM scan.  grid = (D/256, NCHUNK, B); one thread per channel d.
// Pass 1 (!FINAL): local scan with h0=0, emit carry state + sum(delta).
// Pass 3 ( FINAL): scan with corrected h0, emit h2 = h*(1+Dp) + sum_n s*C.
template <bool FINAL>
__global__ __launch_bounds__(256) void scan_pass(
    const float* __restrict__ hin, const float* __restrict__ delta,
    const float* __restrict__ dbc, const float* __restrict__ A_log,
    const float* __restrict__ Dp, const float* __restrict__ h0,
    float* __restrict__ carry_h, float* __restrict__ dsum_out,
    float* __restrict__ h2) {
    __shared__ float bc[32][32];            // 32 timesteps x (16 B | 16 C)
    const int tid = threadIdx.x;
    const int d   = blockIdx.x * 256 + tid;
    const int c   = blockIdx.y;
    const int b   = blockIdx.z;

    float An[N_], s[N_];
#pragma unroll
    for (int n = 0; n < N_; n++) An[n] = -__expf(A_log[(size_t)d * N_ + n]);
    if (FINAL) {
        const float* hp = h0 + ((size_t)(b * NCHUNK + c) * D_ + d) * N_;
#pragma unroll
        for (int n = 0; n < N_; n++) s[n] = hp[n];
    } else {
#pragma unroll
        for (int n = 0; n < N_; n++) s[n] = 0.f;
    }
    const float dpv = FINAL ? Dp[d] : 0.f;
    float dsum = 0.f;
    const int l0 = c * LCHUNK;

    for (int sub = 0; sub < LCHUNK; sub += 32) {
        __syncthreads();
#pragma unroll
        for (int j = 0; j < 4; j++) {       // stage B,C rows for 32 timesteps
            int idx = tid + j * 256;
            int li = idx >> 5, n = idx & 31;
            bc[li][n] = dbc[(size_t)(b * L_ + l0 + sub + li) * E_ + R_ + n];
        }
        __syncthreads();
        for (int li = 0; li < 32; li++) {
            size_t o = ((size_t)(b * L_) + (l0 + sub + li)) * D_ + d;
            float dl = delta[o];
            float xv = hin[o];
            float dx = dl * xv;
            if (!FINAL) dsum += dl;
            float y = 0.f;
#pragma unroll
            for (int n = 0; n < N_; n++) {
                float dA = __expf(dl * An[n]);
                s[n] = dA * s[n] + dx * bc[li][n];
                if (FINAL) y += s[n] * bc[li][16 + n];
            }
            if (FINAL) h2[o] = xv * (1.f + dpv) + y;
        }
    }
    if (!FINAL) {
        float* cp = carry_h + ((size_t)(b * NCHUNK + c) * D_ + d) * N_;
#pragma unroll
        for (int n = 0; n < N_; n++) cp[n] = s[n];
        dsum_out[(size_t)(b * NCHUNK + c) * D_ + d] = dsum;
    }
}

// Pass 2: combine the 8 chunk carries per (b,d).  prod(exp(dl*A)) = exp(A*sum dl).
__global__ __launch_bounds__(256) void scan_carry(
    const float* __restrict__ carry_h, const float* __restrict__ dsum,
    const float* __restrict__ A_log, float* __restrict__ h0) {
    int t = blockIdx.x * 256 + threadIdx.x;   // 0 .. B*D-1
    int b = t >> 10;
    int d = t & (D_ - 1);
    float An[N_], run[N_];
#pragma unroll
    for (int n = 0; n < N_; n++) {
        An[n] = -__expf(A_log[(size_t)d * N_ + n]);
        run[n] = 0.f;
    }
    for (int c = 0; c < NCHUNK; c++) {
        size_t base = (size_t)(b * NCHUNK + c) * D_ + d;
        float* hp = h0 + base * N_;
#pragma unroll
        for (int n = 0; n < N_; n++) hp[n] = run[n];
        float ds = dsum[base];
        const float* cp = carry_h + base * N_;
#pragma unroll
        for (int n = 0; n < N_; n++) run[n] = __expf(An[n] * ds) * run[n] + cp[n];
    }
}

// ------------------------------- launcher ----------------------------------

extern "C" void kernel_launch(void* const* d_in, const int* in_sizes, int n_in,
                              void* d_out, int out_size, void* d_ws, size_t ws_size,
                              hipStream_t stream) {
    (void)in_sizes; (void)n_in; (void)out_size; (void)ws_size;
    const float* x      = (const float*)d_in[0];
    const float* ln1w   = (const float*)d_in[1];
    const float* ln2w   = (const float*)d_in[2];
    const float* W_dbc  = (const float*)d_in[3];
    const float* W_dt   = (const float*)d_in[4];
    const float* b_dt   = (const float*)d_in[5];
    const float* A_log  = (const float*)d_in[6];
    const float* Dp     = (const float*)d_in[7];
    const float* W_fc   = (const float*)d_in[8];
    const float* W_proj = (const float*)d_in[9];

    char* ws = (char*)d_ws;
    size_t off = 0;
    auto alloc = [&](size_t bytes) -> char* {
        char* p = ws + off;
        off += (bytes + 255) & ~(size_t)255;
        return p;
    };
    float*    h32     = (float*)   alloc((size_t)M_ * D_ * 4);
    _Float16* h16     = (_Float16*)alloc((size_t)M_ * D_ * 2);
    float*    dbc32   = (float*)   alloc((size_t)M_ * E_ * 4);
    _Float16* dbc16   = (_Float16*)alloc((size_t)M_ * E_ * 2);
    float*    dlt32   = (float*)   alloc((size_t)M_ * D_ * 4);
    float*    h2      = (float*)   alloc((size_t)M_ * D_ * 4);
    _Float16* m16     = (_Float16*)alloc((size_t)M_ * D_ * 2);
    _Float16* fc16    = (_Float16*)alloc((size_t)M_ * F_ * 2);
    _Float16* wdbc16  = (_Float16*)alloc((size_t)E_ * D_ * 2);
    _Float16* wdt16   = (_Float16*)alloc((size_t)D_ * R_ * 2);
    _Float16* wfc16   = (_Float16*)alloc((size_t)F_ * D_ * 2);
    _Float16* wproj16 = (_Float16*)alloc((size_t)D_ * F_ * 2);
    float*    carryh  = (float*)   alloc((size_t)B_ * NCHUNK * D_ * N_ * 4);
    float*    dsum    = (float*)   alloc((size_t)B_ * NCHUNK * D_ * 4);
    float*    h0buf   = (float*)   alloc((size_t)B_ * NCHUNK * D_ * N_ * 4);

    // weight conversion f32 -> f16
    cvt_f16<<<(E_ * D_ + 255) / 256, 256, 0, stream>>>(W_dbc, wdbc16, E_ * D_);
    cvt_f16<<<(D_ * R_ + 255) / 256, 256, 0, stream>>>(W_dt, wdt16, D_ * R_);
    cvt_f16<<<(F_ * D_ + 255) / 256, 256, 0, stream>>>(W_fc, wfc16, F_ * D_);
    cvt_f16<<<(D_ * F_ + 255) / 256, 256, 0, stream>>>(W_proj, wproj16, D_ * F_);

    // h = softplus(ln1(x))
    ln_kernel<0><<<M_, 256, 0, stream>>>(x, ln1w, h32, h16);

    // dbc = h @ W_dbc^T   (M=4096, N=96, K=1024)
    gemm_wmma<0><<<dim3(1, M_ / 128), 256, 0, stream>>>(
        h16, D_, wdbc16, D_, M_, E_, D_, nullptr, nullptr, dbc32, dbc16, E_);

    // delta = softplus(dbc[:, :R] @ W_dt^T + b_dt)   (M=4096, N=1024, K=64)
    gemm_wmma<1><<<dim3(D_ / 128, M_ / 128), 256, 0, stream>>>(
        dbc16, E_, wdt16, R_, M_, D_, R_, b_dt, nullptr, dlt32, nullptr, D_);

    // chunked SSM scan (3 passes)
    scan_pass<false><<<dim3(D_ / 256, NCHUNK, B_), 256, 0, stream>>>(
        h32, dlt32, dbc32, A_log, Dp, nullptr, carryh, dsum, nullptr);
    scan_carry<<<(B_ * D_) / 256, 256, 0, stream>>>(carryh, dsum, A_log, h0buf);
    scan_pass<true><<<dim3(D_ / 256, NCHUNK, B_), 256, 0, stream>>>(
        h32, dlt32, dbc32, A_log, Dp, h0buf, nullptr, nullptr, h2);

    // m = ln2(h2)
    ln_kernel<1><<<M_, 256, 0, stream>>>(h2, ln2w, nullptr, m16);

    // fc = gelu(m @ W_fc^T)   (M=4096, N=4096, K=1024)
    gemm_wmma<2><<<dim3(F_ / 128, M_ / 128), 256, 0, stream>>>(
        m16, D_, wfc16, D_, M_, F_, D_, nullptr, nullptr, nullptr, fc16, F_);

    // out = h2 + fc @ W_proj^T   (M=4096, N=1024, K=4096)
    gemm_wmma<3><<<dim3(D_ / 128, M_ / 128), 256, 0, stream>>>(
        fc16, F_, wproj16, F_, M_, D_, F_, nullptr, h2, (float*)d_out, nullptr, D_);
}
